// SBaseMoE_4492535791705
// MI455X (gfx1250) — compile-verified, matrix operand order checked by reading, not run on hardware
//
#include <hip/hip_runtime.h>
#include <cstdint>

// ---------------------------------------------------------------------------
// Top-1 MoE for MI455X (gfx1250): route -> gather -> f16 WMMA grouped GEMMs.
// Software-pipelined (double-buffered LDS): A-tiles via Tensor Data Mover
// (tensor_load_to_lds + s_wait_tensorcnt), weights f32->f16 while staging.
// ---------------------------------------------------------------------------

#define HDIM 2048
#define IDIM 4096
#define NEXP 8
#define NTOK 4096                  // B * S
#define MB   64                    // M tile per block
#define NB   64                    // N tile per block
#define KSTEP 32                   // K per WMMA step (16x16x32 f16)
#define KPAD  40                   // padded LDS row (80B: conflict-free, 16B aligned)
#define CAPROWS (NTOK + NEXP * MB) // 64-aligned per-expert padding

#if __has_builtin(__builtin_amdgcn_tensor_load_to_lds)
#define USE_TDM 1
#else
#define USE_TDM 0
#endif

typedef __attribute__((ext_vector_type(16))) _Float16     v16h;
typedef __attribute__((ext_vector_type(8)))  _Float16     v8h;
typedef __attribute__((ext_vector_type(8)))  float        v8f;
typedef __attribute__((ext_vector_type(4)))  unsigned int v4u;
typedef __attribute__((ext_vector_type(8)))  int          v8i;
typedef __attribute__((ext_vector_type(4)))  int          v4i;

// Per-lane 16-bit fragment: elements [0..7] at p, [8..15] at p+16 halves
// (CDNA5 16-bit A layout with p = row_base + (lane>>4)*8).
__device__ __forceinline__ v16h load_frag16(const _Float16* p) {
  v8h lo = *(const v8h*)(p);
  v8h hi = *(const v8h*)(p + 16);
  return __builtin_shufflevector(lo, hi, 0, 1, 2, 3, 4, 5, 6, 7,
                                 8, 9, 10, 11, 12, 13, 14, 15);
}

// TDM: async load of a KSTEP x MB f16 tile (row stride = stride_elems) into
// LDS at lds_off, padding each 64B row to 80B (KPAD layout). D# per CDNA5 ISA
// 8.3/8.4 (group0: count/lds/global/type=2; group1: data_size/pad/dims/stride).
__device__ __forceinline__ void tdm_load_tile_f16(uint32_t lds_off,
                                                  const _Float16* gptr,
                                                  uint32_t stride_elems) {
#if USE_TDM
  uint64_t ga = (uint64_t)(uintptr_t)gptr;
  v4u g0;
  g0[0] = 1u;                                                // count=1 (valid)
  g0[1] = lds_off;                                           // lds_addr (bytes)
  g0[2] = (uint32_t)ga;                                      // global_addr[31:0]
  g0[3] = (uint32_t)((ga >> 32) & 0x1ffffffu) | (2u << 30);  // addr[56:32]|type=2
  v8i g1;
  g1[0] = (int)((1u << 16)    // data_size = 2 bytes
              | (1u << 20)    // pad_enable
              | (3u << 22)    // pad_interval: 16 dwords (64B = one tile row)
              | (3u << 25));  // pad_amount: 4 dwords (16B -> 80B rows)
  g1[1] = (int)0xffff0000u;                    // tensor_dim0 = 0x7fffffff (lo16)
  g1[2] = (int)0xffff7fffu;                    // tensor_dim0 hi | tensor_dim1 lo
  g1[3] = (int)(((unsigned)KSTEP << 16) | 0x7fffu); // tile_dim0 | tensor_dim1 hi
  g1[4] = MB;                                  // tile_dim1 = MB (tile_dim2 = 0)
  g1[5] = (int)stride_elems;                   // tensor_dim0_stride lo32
  g1[6] = 0;
  g1[7] = 0;
  v4i z = {0, 0, 0, 0};
#if __has_include(<hip/amd_detail/amd_gfx1250_TDM.h>)
  v8i z8 = {0, 0, 0, 0, 0, 0, 0, 0};
  __builtin_amdgcn_tensor_load_to_lds(g0, g1, z, z, z8, 0);
#else
  __builtin_amdgcn_tensor_load_to_lds(g0, g1, z, z, 0);
#endif
#else
  (void)lds_off; (void)gptr; (void)stride_elems;
#endif
}

__global__ void k_zero(int* counts) {
  if (threadIdx.x < NEXP) counts[threadIdx.x] = 0;
}

// One thread per token: 8 router dot products, argmax, sigmoid, slot assign.
__global__ void k_router(const float* __restrict__ x, const float* __restrict__ rw,
                         int* __restrict__ expert, int* __restrict__ slot,
                         float* __restrict__ prob, int* __restrict__ counts) {
  int t = blockIdx.x * blockDim.x + threadIdx.x;
  if (t >= NTOK) return;
  const float* xr = x + (size_t)t * HDIM;
  float acc[NEXP] = {};
  for (int h = 0; h < HDIM; ++h) {
    float xv = xr[h];
    const float* r = rw + h * NEXP;  // wave-uniform -> scalar loads
#pragma unroll
    for (int e = 0; e < NEXP; ++e) acc[e] = __builtin_fmaf(xv, r[e], acc[e]);
  }
  int best = 0;
  float bm = acc[0];
#pragma unroll
  for (int e = 1; e < NEXP; ++e)
    if (acc[e] > bm) { bm = acc[e]; best = e; }  // first-max tie break
  expert[t] = best;
  prob[t] = 1.0f / (1.0f + __expf(-bm));
  slot[t] = atomicAdd(&counts[best], 1);
}

// 64-aligned exclusive scan of the 8 counts.
__global__ void k_offsets(const int* __restrict__ counts, int* __restrict__ offs) {
  if (threadIdx.x == 0) {
    int o = 0;
    for (int e = 0; e < NEXP; ++e) {
      offs[e] = o;
      o += (counts[e] + MB - 1) & ~(MB - 1);
    }
  }
}

// One block per token: convert+copy its H row into the gathered f16 buffer.
__global__ void k_gather(const float* __restrict__ x, _Float16* __restrict__ Xg,
                         int* __restrict__ toksort, const int* __restrict__ expert,
                         const int* __restrict__ slot, const int* __restrict__ offs) {
  int t = blockIdx.x;
  int e = expert[t];
  int row = offs[e] + slot[t];
  if (threadIdx.x == 0) toksort[row] = t;
  int c = threadIdx.x * 8;  // 256 threads * 8 = 2048 = HDIM
  const float4* src = (const float4*)(x + (size_t)t * HDIM + c);
  float4 a = src[0], b = src[1];
  v8h o;
  o[0] = (_Float16)a.x; o[1] = (_Float16)a.y; o[2] = (_Float16)a.z; o[3] = (_Float16)a.w;
  o[4] = (_Float16)b.x; o[5] = (_Float16)b.y; o[6] = (_Float16)b.z; o[7] = (_Float16)b.w;
  *(v8h*)(Xg + (size_t)row * HDIM + c) = o;
}

// GEMM1: h = silu(Xe @ gate) * (Xe @ up). Block tile 64x64, 8 waves; each wave:
// one 16-row M-subtile x two 16-col N-subtiles for gate and up (4 WMMA/step).
// 2-deep pipeline: stage k+1 (TDM A + b128 weight loads) under the WMMAs of k.
__global__ void __launch_bounds__(256)
k_gemm1(const _Float16* __restrict__ Xg, const float* __restrict__ gate_w,
        const float* __restrict__ up_w, _Float16* __restrict__ Hws,
        const int* __restrict__ counts, const int* __restrict__ offs) {
  int e = blockIdx.z;
  int ne = counts[e];
  int m_base = blockIdx.y * MB;
  if (m_base >= ne) return;
  int n0 = blockIdx.x * NB;
  const float* Wg = gate_w + (size_t)e * HDIM * IDIM;
  const float* Wu = up_w + (size_t)e * HDIM * IDIM;
  int rowbase = offs[e] + m_base;
  const _Float16* aSrc = Xg + (size_t)rowbase * HDIM;

  __shared__ alignas(16) _Float16 lA[2][MB * KPAD];    // [m][k], 80B rows
  __shared__ alignas(16) _Float16 lBg[2][NB * KPAD];   // [n][k] (transposed)
  __shared__ alignas(16) _Float16 lBu[2][NB * KPAD];

  int tid = threadIdx.x;
  int lane = tid & 31, wave = tid >> 5;
  int m_sub = wave >> 1, np = wave & 1;  // 4 M-subtiles x 2 N-pair groups
  int r = lane & 15, hf = lane >> 4;
  int bk = tid >> 4;                     // 0..15 (k within step)
  int bn = (tid & 15) * 4;               // 0..60 (n within tile)
  v8f accG0 = {}, accG1 = {}, accU0 = {}, accU1 = {};

  // ---- prologue: stage step 0 into buffer 0
  {
#if USE_TDM
    if (wave == 0) tdm_load_tile_f16((uint32_t)(uintptr_t)&lA[0][0], aSrc, HDIM);
#else
    int rr = tid >> 2, cc = (tid & 3) * 8;
    *(uint4*)(&lA[0][0] + rr * KPAD + cc) =
        *(const uint4*)(aSrc + (size_t)rr * HDIM + cc);
#endif
#pragma unroll
    for (int q = 0; q < 2; ++q) {
      int kk = bk + q * 16;
      size_t gi = (size_t)kk * IDIM + n0 + bn;
      float4 wg = *(const float4*)(Wg + gi);
      float4 wu = *(const float4*)(Wu + gi);
      lBg[0][(bn + 0) * KPAD + kk] = (_Float16)wg.x;
      lBg[0][(bn + 1) * KPAD + kk] = (_Float16)wg.y;
      lBg[0][(bn + 2) * KPAD + kk] = (_Float16)wg.z;
      lBg[0][(bn + 3) * KPAD + kk] = (_Float16)wg.w;
      lBu[0][(bn + 0) * KPAD + kk] = (_Float16)wu.x;
      lBu[0][(bn + 1) * KPAD + kk] = (_Float16)wu.y;
      lBu[0][(bn + 2) * KPAD + kk] = (_Float16)wu.z;
      lBu[0][(bn + 3) * KPAD + kk] = (_Float16)wu.w;
    }
#if USE_TDM
    if (wave == 0) __builtin_amdgcn_s_wait_tensorcnt(0);
#endif
    __syncthreads();
  }

  const int NSTEP = HDIM / KSTEP;
  for (int ks = 0; ks < NSTEP; ++ks) {
    int buf = ks & 1, nxt = buf ^ 1;
    bool more = (ks + 1) < NSTEP;
    int k1 = (ks + 1) * KSTEP;

    // issue next-step staging (loads in flight behind the WMMAs below)
    float4 wg0 = {}, wg1 = {}, wu0 = {}, wu1 = {};
    if (more) {
#if USE_TDM
      if (wave == 0)
        tdm_load_tile_f16((uint32_t)(uintptr_t)&lA[nxt][0], aSrc + k1, HDIM);
#else
      int rr = tid >> 2, cc = (tid & 3) * 8;
      *(uint4*)(&lA[nxt][0] + rr * KPAD + cc) =
          *(const uint4*)(aSrc + (size_t)rr * HDIM + k1 + cc);
#endif
      size_t gi0 = (size_t)(k1 + bk) * IDIM + n0 + bn;
      size_t gi1 = (size_t)(k1 + bk + 16) * IDIM + n0 + bn;
      wg0 = *(const float4*)(Wg + gi0);
      wg1 = *(const float4*)(Wg + gi1);
      wu0 = *(const float4*)(Wu + gi0);
      wu1 = *(const float4*)(Wu + gi1);
    }

    // compute current buffer
    const _Float16* A  = &lA[buf][0];
    const _Float16* Bg = &lBg[buf][0];
    const _Float16* Bu = &lBu[buf][0];
    v16h a = load_frag16(A + (m_sub * 16 + r) * KPAD + hf * 8);
    const _Float16* pg = Bg + (np * 32 + r) * KPAD + hf * 8;
    const _Float16* pu = Bu + (np * 32 + r) * KPAD + hf * 8;
    v16h bg0 = load_frag16(pg);
    v16h bg1 = load_frag16(pg + 16 * KPAD);
    v16h bu0 = load_frag16(pu);
    v16h bu1 = load_frag16(pu + 16 * KPAD);
    accG0 = __builtin_amdgcn_wmma_f32_16x16x32_f16(false, a, false, bg0,
                                                   (short)0, accG0, false, false);
    accG1 = __builtin_amdgcn_wmma_f32_16x16x32_f16(false, a, false, bg1,
                                                   (short)0, accG1, false, false);
    accU0 = __builtin_amdgcn_wmma_f32_16x16x32_f16(false, a, false, bu0,
                                                   (short)0, accU0, false, false);
    accU1 = __builtin_amdgcn_wmma_f32_16x16x32_f16(false, a, false, bu1,
                                                   (short)0, accU1, false, false);

    // convert + store next-step B tiles
    if (more) {
      _Float16* dg = &lBg[nxt][0];
      _Float16* du = &lBu[nxt][0];
      int kka = bk, kkb = bk + 16;
      dg[(bn + 0) * KPAD + kka] = (_Float16)wg0.x;
      dg[(bn + 1) * KPAD + kka] = (_Float16)wg0.y;
      dg[(bn + 2) * KPAD + kka] = (_Float16)wg0.z;
      dg[(bn + 3) * KPAD + kka] = (_Float16)wg0.w;
      dg[(bn + 0) * KPAD + kkb] = (_Float16)wg1.x;
      dg[(bn + 1) * KPAD + kkb] = (_Float16)wg1.y;
      dg[(bn + 2) * KPAD + kkb] = (_Float16)wg1.z;
      dg[(bn + 3) * KPAD + kkb] = (_Float16)wg1.w;
      du[(bn + 0) * KPAD + kka] = (_Float16)wu0.x;
      du[(bn + 1) * KPAD + kka] = (_Float16)wu0.y;
      du[(bn + 2) * KPAD + kka] = (_Float16)wu0.z;
      du[(bn + 3) * KPAD + kka] = (_Float16)wu0.w;
      du[(bn + 0) * KPAD + kkb] = (_Float16)wu1.x;
      du[(bn + 1) * KPAD + kkb] = (_Float16)wu1.y;
      du[(bn + 2) * KPAD + kkb] = (_Float16)wu1.z;
      du[(bn + 3) * KPAD + kkb] = (_Float16)wu1.w;
#if USE_TDM
      if (wave == 0) __builtin_amdgcn_s_wait_tensorcnt(0);
#endif
    }
    __syncthreads();  // one barrier per k-step
  }

  // Epilogue: silu(g)*u -> f16. C layout: VGPR j, lanes 0-15: (M=j, N=lane);
  // lanes 16-31: (M=j+8, N=lane-16).
#pragma unroll
  for (int j = 0; j < 8; ++j) {
    int m = j + hf * 8;
    size_t grow = (size_t)(rowbase + m_sub * 16 + m);
#pragma unroll
    for (int s = 0; s < 2; ++s) {
      float g = s ? accG1[j] : accG0[j];
      float u = s ? accU1[j] : accU0[j];
      float hv = (g / (1.0f + __expf(-g))) * u;
      Hws[grow * IDIM + n0 + np * 32 + s * 16 + r] = (_Float16)hv;
    }
  }
}

// GEMM2: out[token] = (h_e @ down_e) * prob[token], scattered; same pipeline.
__global__ void __launch_bounds__(256)
k_gemm2(const _Float16* __restrict__ Hws, const float* __restrict__ down_w,
        float* __restrict__ out, const float* __restrict__ prob,
        const int* __restrict__ toksort, const int* __restrict__ counts,
        const int* __restrict__ offs) {
  int e = blockIdx.z;
  int ne = counts[e];
  int m_base = blockIdx.y * MB;
  if (m_base >= ne) return;
  int n0 = blockIdx.x * NB;
  const float* Wd = down_w + (size_t)e * IDIM * HDIM;
  int rowbase = offs[e] + m_base;
  const _Float16* aSrc = Hws + (size_t)rowbase * IDIM;

  __shared__ alignas(16) _Float16 lA[2][MB * KPAD];
  __shared__ alignas(16) _Float16 lBd[2][NB * KPAD];

  int tid = threadIdx.x;
  int lane = tid & 31, wave = tid >> 5;
  int m_sub = wave >> 1, np = wave & 1;
  int r = lane & 15, hf = lane >> 4;
  int bk = tid >> 4;
  int bn = (tid & 15) * 4;
  v8f acc0 = {}, acc1 = {};

  {  // prologue
#if USE_TDM
    if (wave == 0) tdm_load_tile_f16((uint32_t)(uintptr_t)&lA[0][0], aSrc, IDIM);
#else
    int rr = tid >> 2, cc = (tid & 3) * 8;
    *(uint4*)(&lA[0][0] + rr * KPAD + cc) =
        *(const uint4*)(aSrc + (size_t)rr * IDIM + cc);
#endif
#pragma unroll
    for (int q = 0; q < 2; ++q) {
      int kk = bk + q * 16;
      float4 wd = *(const float4*)(Wd + (size_t)kk * HDIM + n0 + bn);
      lBd[0][(bn + 0) * KPAD + kk] = (_Float16)wd.x;
      lBd[0][(bn + 1) * KPAD + kk] = (_Float16)wd.y;
      lBd[0][(bn + 2) * KPAD + kk] = (_Float16)wd.z;
      lBd[0][(bn + 3) * KPAD + kk] = (_Float16)wd.w;
    }
#if USE_TDM
    if (wave == 0) __builtin_amdgcn_s_wait_tensorcnt(0);
#endif
    __syncthreads();
  }

  const int NSTEP = IDIM / KSTEP;
  for (int ks = 0; ks < NSTEP; ++ks) {
    int buf = ks & 1, nxt = buf ^ 1;
    bool more = (ks + 1) < NSTEP;
    int k1 = (ks + 1) * KSTEP;

    float4 wd0 = {}, wd1 = {};
    if (more) {
#if USE_TDM
      if (wave == 0)
        tdm_load_tile_f16((uint32_t)(uintptr_t)&lA[nxt][0], aSrc + k1, IDIM);
#else
      int rr = tid >> 2, cc = (tid & 3) * 8;
      *(uint4*)(&lA[nxt][0] + rr * KPAD + cc) =
          *(const uint4*)(aSrc + (size_t)rr * IDIM + k1 + cc);
#endif
      wd0 = *(const float4*)(Wd + (size_t)(k1 + bk) * HDIM + n0 + bn);
      wd1 = *(const float4*)(Wd + (size_t)(k1 + bk + 16) * HDIM + n0 + bn);
    }

    const _Float16* A  = &lA[buf][0];
    const _Float16* Bd = &lBd[buf][0];
    v16h a = load_frag16(A + (m_sub * 16 + r) * KPAD + hf * 8);
    const _Float16* pd = Bd + (np * 32 + r) * KPAD + hf * 8;
    v16h b0 = load_frag16(pd);
    v16h b1 = load_frag16(pd + 16 * KPAD);
    acc0 = __builtin_amdgcn_wmma_f32_16x16x32_f16(false, a, false, b0,
                                                  (short)0, acc0, false, false);
    acc1 = __builtin_amdgcn_wmma_f32_16x16x32_f16(false, a, false, b1,
                                                  (short)0, acc1, false, false);

    if (more) {
      _Float16* dd = &lBd[nxt][0];
      int kka = bk, kkb = bk + 16;
      dd[(bn + 0) * KPAD + kka] = (_Float16)wd0.x;
      dd[(bn + 1) * KPAD + kka] = (_Float16)wd0.y;
      dd[(bn + 2) * KPAD + kka] = (_Float16)wd0.z;
      dd[(bn + 3) * KPAD + kka] = (_Float16)wd0.w;
      dd[(bn + 0) * KPAD + kkb] = (_Float16)wd1.x;
      dd[(bn + 1) * KPAD + kkb] = (_Float16)wd1.y;
      dd[(bn + 2) * KPAD + kkb] = (_Float16)wd1.z;
      dd[(bn + 3) * KPAD + kkb] = (_Float16)wd1.w;
#if USE_TDM
      if (wave == 0) __builtin_amdgcn_s_wait_tensorcnt(0);
#endif
    }
    __syncthreads();
  }

#pragma unroll
  for (int j = 0; j < 8; ++j) {
    int m = m_sub * 16 + j + hf * 8;
    if (m_base + m < ne) {  // skip padded rows
      int tok = toksort[rowbase + m];
      float p = prob[tok];
#pragma unroll
      for (int s = 0; s < 2; ++s) {
        float v = s ? acc1[j] : acc0[j];
        out[(size_t)tok * HDIM + n0 + np * 32 + s * 16 + r] = v * p;
      }
    }
  }
}

extern "C" void kernel_launch(void* const* d_in, const int* in_sizes, int n_in,
                              void* d_out, int out_size, void* d_ws, size_t ws_size,
                              hipStream_t stream) {
  const float* x  = (const float*)d_in[0];  // [B,S,H]
  const float* rw = (const float*)d_in[1];  // [H,E]
  const float* gw = (const float*)d_in[2];  // [E,H,I]
  const float* uw = (const float*)d_in[3];  // [E,H,I]
  const float* dw = (const float*)d_in[4];  // [E,I,H]
  float* out = (float*)d_out;
  (void)in_sizes; (void)n_in; (void)out_size; (void)ws_size;

  char* ws = (char*)d_ws;
  auto carve = [&](size_t bytes) {
    char* p = ws;
    ws += (bytes + 255) & ~(size_t)255;
    return p;
  };
  int*      counts  = (int*)carve(NEXP * sizeof(int));
  int*      offs    = (int*)carve(NEXP * sizeof(int));
  int*      expert  = (int*)carve(NTOK * sizeof(int));
  int*      slot    = (int*)carve(NTOK * sizeof(int));
  float*    prob    = (float*)carve(NTOK * sizeof(float));
  int*      toksort = (int*)carve(CAPROWS * sizeof(int));
  _Float16* Xg      = (_Float16*)carve((size_t)CAPROWS * HDIM * sizeof(_Float16));
  _Float16* Hws     = (_Float16*)carve((size_t)CAPROWS * IDIM * sizeof(_Float16));

  k_zero<<<1, 32, 0, stream>>>(counts);
  k_router<<<NTOK / 256, 256, 0, stream>>>(x, rw, expert, slot, prob, counts);
  k_offsets<<<1, 32, 0, stream>>>(counts, offs);
  k_gather<<<NTOK, 256, 0, stream>>>(x, Xg, toksort, expert, slot, offs);

  dim3 g1(IDIM / NB, NTOK / MB, NEXP);
  k_gemm1<<<g1, 256, 0, stream>>>(Xg, gw, uw, Hws, counts, offs);
  dim3 g2(HDIM / NB, NTOK / MB, NEXP);
  k_gemm2<<<g2, 256, 0, stream>>>(Hws, dw, out, prob, toksort, counts, offs);
}
